// MNISTGraphNN_55714315764269
// MI455X (gfx1250) — matrix operand order repaired and checked by reading, not compiled, not run on hardware
//
#include <hip/hip_runtime.h>
#include <hip/hip_bf16.h>

// ---------------------------------------------------------------------------
// MNIST GraphNN forward on gfx1250 (MI455X). WMMA f16 (fp32 acc) for all
// heavy GEMMs; aggregation done in fin-space via softmax-linearity trick.
// ---------------------------------------------------------------------------

typedef __attribute__((ext_vector_type(16))) _Float16 v16h;
typedef __attribute__((ext_vector_type(8)))  float    v8f;

#define IMGX   28
#define NNODE  784
#define BATCH  128
#define MTOT   (BATCH * NNODE)   // 100352, multiple of 16
#define LN_EPS 1e-5f

// ---------------------------------------------------------------------------
// h0 builder: h0[b,n,0..2] = (pix, xc, yc), padded with zeros to 32 cols.
// ---------------------------------------------------------------------------
__global__ void gnn_build_h0(const float* __restrict__ x, _Float16* __restrict__ h0)
{
    int node = blockIdx.x * blockDim.x + threadIdx.x;
    if (node >= MTOT) return;
    int n = node % NNODE;
    int i = n / IMGX, j = n % IMGX;
    float pix = x[node];
    float xc = (float)(j - 14) * (1.0f / 14.0f);
    float yc = (float)(i - 14) * (1.0f / 14.0f);
    _Float16* p = h0 + (size_t)node * 32;
    p[0] = (_Float16)pix;
    p[1] = (_Float16)xc;
    p[2] = (_Float16)yc;
    #pragma unroll
    for (int k = 3; k < 32; ++k) p[k] = (_Float16)0.0f;
}

// ---------------------------------------------------------------------------
// Pack fp32 weight [FINr, FOUT] (row-major) into the WMMA B-operand layout:
// per (kt, nt) tile of 32 lanes x 16 halves:
//   lane l holds column n = nt*16 + (l&15), K values k = kt*32 + (l>=16?8:0) + (h&7) + (h>=8?16:0)
// Zero-pads K beyond FINr (layer 0: FINr=3 padded to 32).
// ---------------------------------------------------------------------------
__global__ void gnn_pack_w(const float* __restrict__ W, _Float16* __restrict__ dst,
                           int FINr, int FINP, int FOUT)
{
    int t = blockIdx.x * blockDim.x + threadIdx.x;
    int total = FINP * FOUT;
    if (t >= total) return;
    int NT = FOUT >> 4;
    int chunk = t >> 9;           // 512 halves per (kt,nt) tile
    int rem   = t & 511;
    int lane  = rem >> 4;
    int hh    = rem & 15;
    int kt = chunk / NT, nt = chunk % NT;
    int n = nt * 16 + (lane & 15);
    int k = kt * 32 + ((lane >= 16) ? 8 : 0) + (hh & 7) + ((hh & 8) ? 16 : 0);
    float v = (k < FINr) ? W[k * FOUT + n] : 0.0f;
    dst[t] = (_Float16)v;
}

// Pack a_j half of W_att ([2*fin,1]) to f16, zero-padded to FINP.
__global__ void gnn_pack_att(const float* __restrict__ Wa, _Float16* __restrict__ dst,
                             int FINr, int FINP)
{
    int t = blockIdx.x * blockDim.x + threadIdx.x;
    if (t >= FINP) return;
    dst[t] = (_Float16)((t < FINr) ? Wa[FINr + t] : 0.0f);
}

__global__ void gnn_combine_bias(const float* __restrict__ bs, const float* __restrict__ bn,
                                 float* __restrict__ dst, int FOUT)
{
    int t = blockIdx.x * blockDim.x + threadIdx.x;
    if (t < FOUT) dst[t] = bs[t] + bn[t];
}

// ---------------------------------------------------------------------------
// a_j[node] = dot(h[node,:], wa_j).  (a_i and b_att cancel in the softmax.)
// ---------------------------------------------------------------------------
__global__ void gnn_att_score(const _Float16* __restrict__ h, const _Float16* __restrict__ waj,
                              float* __restrict__ aj, int FINP)
{
    int node = blockIdx.x * blockDim.x + threadIdx.x;
    if (node >= MTOT) return;
    const _Float16* hp = h + (size_t)node * FINP;
    float s = 0.0f;
    for (int k = 0; k < FINP; ++k) s += (float)hp[k] * (float)waj[k];
    aj[node] = s;
}

// ---------------------------------------------------------------------------
// Aggregate: one wave per node.  Neighbor structure of the 28x28 grid with
// radius-2 (12 offsets) is computed on the fly.  hagg = softmax-weighted sum
// of neighbor features (fin-space).
// ---------------------------------------------------------------------------
__global__ __launch_bounds__(128)
void gnn_aggregate(const _Float16* __restrict__ h, const float* __restrict__ aj,
                   _Float16* __restrict__ hagg, int FINP)
{
    int wave = blockIdx.x * 4 + (threadIdx.x >> 5);
    int lane = threadIdx.x & 31;
    int b = wave / NNODE, n = wave % NNODE;
    int i = n / IMGX, j = n % IMGX;

    const int di[12] = {-2,-1,-1,-1, 0, 0, 0, 0, 1, 1, 1, 2};
    const int dj[12] = { 0,-1, 0, 1,-2,-1, 1, 2,-1, 0, 1, 0};

    float s[12]; int idx[12]; bool val[12];
    float mx = -3.0e38f;
    #pragma unroll
    for (int k = 0; k < 12; ++k) {
        int ni = i + di[k], nj = j + dj[k];
        bool v = (ni >= 0) & (ni < IMGX) & (nj >= 0) & (nj < IMGX);
        val[k] = v;
        idx[k] = v ? (b * NNODE + ni * IMGX + nj) : wave;  // self if invalid (weight=0)
        float a = aj[idx[k]];
        if (!v) a = -3.0e38f;
        s[k] = a;
        mx = fmaxf(mx, a);
    }
    float w[12]; float wsum = 0.0f;
    #pragma unroll
    for (int k = 0; k < 12; ++k) {
        float e = val[k] ? __expf(s[k] - mx) : 0.0f;
        w[k] = e; wsum += e;
    }
    float inv = 1.0f / wsum;   // every grid node has >=1 neighbor
    #pragma unroll
    for (int k = 0; k < 12; ++k) w[k] *= inv;

    for (int f = lane; f < FINP; f += 32) {
        float a = 0.0f;
        #pragma unroll
        for (int k = 0; k < 12; ++k)
            a += w[k] * (float)h[(size_t)idx[k] * FINP + f];
        hagg[(size_t)wave * FINP + f] = (_Float16)a;
    }
}

// ---------------------------------------------------------------------------
// WMMA GEMM pass: acc[NT] += A(16x32 f16 tile rows) x Bpacked(kt,nt).
// A layout (16-bit A 16x32): lane L: row M=L&15;
//   halves 0..7  = K (kt*32 + khalf .. +7),  halves 8..15 = K (+16 .. +23),
//   khalf = 0 for lanes 0-15, 8 for lanes 16-31.
// ---------------------------------------------------------------------------
template <int FINP, int NT>
__device__ inline void gnn_gemm_pass(const _Float16* __restrict__ A,
                                     const _Float16* __restrict__ Bp,
                                     int lane, int khalf, v8f (&acc)[NT])
{
    constexpr int KT = FINP / 32;
    #pragma unroll
    for (int kt = 0; kt < KT; ++kt) {
        union { v16h v; uint4 u[2]; } af;
        const _Float16* ap = A + kt * 32 + khalf;
        af.u[0] = *(const uint4*)(ap);
        af.u[1] = *(const uint4*)(ap + 16);
        #pragma unroll
        for (int nt = 0; nt < NT; ++nt) {
            v16h bf = *(const v16h*)(Bp + ((size_t)(kt * NT + nt) * 32 + lane) * 16);
            acc[nt] = __builtin_amdgcn_wmma_f32_16x16x32_f16(
                false, af.v, false, bf, (short)0, acc[nt], false, false);
        }
    }
}

// ---------------------------------------------------------------------------
// Fused layer GEMM: c = h@Ws + hagg@Wn + (bs+bn); LayerNorm over FOUT; ReLU.
// One wave owns a 16-row M-tile x full FOUT.  C/D layout: VGPR r holds row
// (mbase + r + 8*(lane>=16)), col = nt*16 + (lane&15); per-row LN stats via
// shfl_xor masks 1/2/4/8 (reduce within each 16-lane half).
// ---------------------------------------------------------------------------
template <int FINP, int FOUT>
__global__ __launch_bounds__(256)
void gnn_gemm_ln(const _Float16* __restrict__ hin, const _Float16* __restrict__ hagg,
                 const _Float16* __restrict__ WsP, const _Float16* __restrict__ WnP,
                 const float* __restrict__ bcomb,
                 const float* __restrict__ lnw, const float* __restrict__ lnb,
                 _Float16* __restrict__ hout)
{
    constexpr int NT = FOUT / 16;
    int wtile = blockIdx.x * 8 + (threadIdx.x >> 5);
    int lane  = threadIdx.x & 31;
    int mbase = wtile * 16;
    int mrow  = mbase + (lane & 15);
    int khalf = (lane & 16) ? 8 : 0;

    v8f acc[NT];
    #pragma unroll
    for (int nt = 0; nt < NT; ++nt) acc[nt] = (v8f){0,0,0,0,0,0,0,0};

    gnn_gemm_pass<FINP, NT>(hin  + (size_t)mrow * FINP, WsP, lane, khalf, acc);
    gnn_gemm_pass<FINP, NT>(hagg + (size_t)mrow * FINP, WnP, lane, khalf, acc);

    // bias + per-row stats (each VGPR slot r is a distinct row)
    int ncol = lane & 15;
    float s1[8], s2[8];
    #pragma unroll
    for (int r = 0; r < 8; ++r) { s1[r] = 0.0f; s2[r] = 0.0f; }
    #pragma unroll
    for (int nt = 0; nt < NT; ++nt) {
        float bb = bcomb[nt * 16 + ncol];
        #pragma unroll
        for (int r = 0; r < 8; ++r) {
            float v = acc[nt][r] + bb;
            acc[nt][r] = v;
            s1[r] += v; s2[r] += v * v;
        }
    }
    #pragma unroll
    for (int r = 0; r < 8; ++r) {
        #pragma unroll
        for (int m = 1; m < 16; m <<= 1) {
            s1[r] += __shfl_xor(s1[r], m, 32);
            s2[r] += __shfl_xor(s2[r], m, 32);
        }
    }
    const float invF = 1.0f / (float)FOUT;
    float mean[8], rs[8];
    #pragma unroll
    for (int r = 0; r < 8; ++r) {
        mean[r] = s1[r] * invF;
        float var = s2[r] * invF - mean[r] * mean[r];
        rs[r] = rsqrtf(var + LN_EPS);
    }
    int rhi = (lane & 16) ? 8 : 0;
    #pragma unroll
    for (int nt = 0; nt < NT; ++nt) {
        int n = nt * 16 + ncol;
        float gw = lnw[n], gb = lnb[n];
        #pragma unroll
        for (int r = 0; r < 8; ++r) {
            float v = (acc[nt][r] - mean[r]) * rs[r] * gw + gb;
            v = v > 0.0f ? v : 0.0f;
            hout[(size_t)(mbase + rhi + r) * FOUT + n] = (_Float16)v;
        }
    }
}

// ---------------------------------------------------------------------------
// Pooling score: sc[node] = relu(h@Wp1+bp1)@Wp2 + bp2.  One wave per node.
// ---------------------------------------------------------------------------
__global__ __launch_bounds__(128)
void gnn_pool_score(const _Float16* __restrict__ h,
                    const float* __restrict__ Wp1, const float* __restrict__ bp1,
                    const float* __restrict__ Wp2, const float* __restrict__ bp2,
                    float* __restrict__ sc)
{
    int wave = blockIdx.x * 4 + (threadIdx.x >> 5);
    int lane = threadIdx.x & 31;
    const _Float16* hp = h + (size_t)wave * 128;
    float part = 0.0f;
    #pragma unroll
    for (int hu = lane; hu < 64; hu += 32) {
        float t = bp1[hu];
        for (int f = 0; f < 128; ++f) t += (float)hp[f] * Wp1[f * 64 + hu];
        t = fmaxf(t, 0.0f);
        part += t * Wp2[hu];
    }
    #pragma unroll
    for (int m = 1; m < 32; m <<= 1) part += __shfl_xor(part, m, 32);
    if (lane == 0) sc[wave] = part + bp2[0];
}

// ---------------------------------------------------------------------------
// Head: per-batch softmax over nodes, pooled rep, classifier.  1 block/batch.
// ---------------------------------------------------------------------------
__global__ __launch_bounds__(256)
void gnn_pool_head(const _Float16* __restrict__ h, const float* __restrict__ sc,
                   const float* __restrict__ Wc1, const float* __restrict__ bc1,
                   const float* __restrict__ Wc2, const float* __restrict__ bc2,
                   float* __restrict__ out)
{
    __shared__ float se[NNODE];
    __shared__ float red[256];
    __shared__ float rep[128];
    __shared__ float c1[64];
    int b = blockIdx.x, t = threadIdx.x;
    const float* scb = sc + (size_t)b * NNODE;

    float m = -3.0e38f;
    for (int n = t; n < NNODE; n += 256) m = fmaxf(m, scb[n]);
    red[t] = m; __syncthreads();
    for (int s = 128; s > 0; s >>= 1) { if (t < s) red[t] = fmaxf(red[t], red[t + s]); __syncthreads(); }
    m = red[0]; __syncthreads();

    float ssum = 0.0f;
    for (int n = t; n < NNODE; n += 256) { float e = __expf(scb[n] - m); se[n] = e; ssum += e; }
    red[t] = ssum; __syncthreads();
    for (int s = 128; s > 0; s >>= 1) { if (t < s) red[t] += red[t + s]; __syncthreads(); }
    float inv = 1.0f / red[0];
    __syncthreads();

    if (t < 128) {
        float a = 0.0f;
        const _Float16* hb = h + ((size_t)b * NNODE) * 128 + t;
        for (int n = 0; n < NNODE; ++n) a += se[n] * (float)hb[(size_t)n * 128];
        rep[t] = a * inv;
    }
    __syncthreads();
    if (t < 64) {
        float a = bc1[t];
        for (int f = 0; f < 128; ++f) a += rep[f] * Wc1[f * 64 + t];
        c1[t] = fmaxf(a, 0.0f);
    }
    __syncthreads();
    if (t < 10) {
        float a = bc2[t];
        for (int hu = 0; hu < 64; ++hu) a += c1[hu] * Wc2[hu * 10 + t];
        out[b * 10 + t] = a;
    }
}

// ---------------------------------------------------------------------------
// Host side
// ---------------------------------------------------------------------------
struct GnnLayerP { const float *Ws, *bs, *Wn, *bn, *Wa, *ba, *lnw, *lnb; };
struct GnnPtrs {
    const float* x;
    GnnLayerP L[4];
    const float *Wp1, *bp1, *Wp2, *bp2, *Wc1, *bc1, *Wc2, *bc2;
};

extern "C" void kernel_launch(void* const* d_in, const int* in_sizes, int n_in,
                              void* d_out, int out_size, void* d_ws, size_t ws_size,
                              hipStream_t stream)
{
    (void)out_size; (void)ws_size;
    GnnPtrs P;
    auto f = [&](int i) { return (const float*)d_in[i]; };

    bool insertion = (n_in >= 1 && in_sizes[0] == MTOT);
    if (insertion) {
        // setup_inputs() dict insertion order: x, then per layer
        // {W_self,b_self,W_nbr,b_nbr,W_att,b_att,ln_w,ln_b}, then head params.
        P.x = f(0);
        int i = 1;
        for (int l = 0; l < 4; ++l) {
            P.L[l].Ws = f(i++); P.L[l].bs = f(i++);
            P.L[l].Wn = f(i++); P.L[l].bn = f(i++);
            P.L[l].Wa = f(i++); P.L[l].ba = f(i++);
            P.L[l].lnw = f(i++); P.L[l].lnb = f(i++);
        }
        P.Wp1 = f(i++); P.bp1 = f(i++); P.Wp2 = f(i++); P.bp2 = f(i++);
        P.Wc1 = f(i++); P.bc1 = f(i++); P.Wc2 = f(i++); P.bc2 = f(i++);
    } else {
        // jax tree_leaves order (dict keys sorted at every level):
        // Wc1,Wc2,Wp1,Wp2,bc1,bc2,bp1,bp2,
        // per layer {W_att,W_nbr,W_self,b_att,b_nbr,b_self,ln_b,ln_w}, then x.
        P.Wc1 = f(0); P.Wc2 = f(1); P.Wp1 = f(2); P.Wp2 = f(3);
        P.bc1 = f(4); P.bc2 = f(5); P.bp1 = f(6); P.bp2 = f(7);
        int i = 8;
        for (int l = 0; l < 4; ++l) {
            P.L[l].Wa = f(i++); P.L[l].Wn = f(i++); P.L[l].Ws = f(i++);
            P.L[l].ba = f(i++); P.L[l].bn = f(i++); P.L[l].bs = f(i++);
            P.L[l].lnb = f(i++); P.L[l].lnw = f(i++);
        }
        P.x = f(40);
    }

    const int FINR[4] = {3, 64, 128, 256};
    const int FINP[4] = {32, 64, 128, 256};
    const int FOUT[4] = {64, 128, 256, 128};

    // Workspace layout (deterministic)
    char* w = (char*)d_ws;
    size_t off = 0;
    auto take = [&](size_t bytes) -> char* {
        char* p = w + off;
        off = (off + bytes + 255) & ~(size_t)255;
        return p;
    };
    _Float16* hA = (_Float16*)take((size_t)MTOT * 256 * 2);
    _Float16* hB = (_Float16*)take((size_t)MTOT * 256 * 2);
    _Float16* hG = (_Float16*)take((size_t)MTOT * 256 * 2);
    float* aj = (float*)take((size_t)MTOT * 4);
    float* sc = (float*)take((size_t)MTOT * 4);
    _Float16 *WsP[4], *WnP[4], *WaP[4];
    float* bcomb[4];
    for (int l = 0; l < 4; ++l) {
        WsP[l] = (_Float16*)take((size_t)FINP[l] * FOUT[l] * 2);
        WnP[l] = (_Float16*)take((size_t)FINP[l] * FOUT[l] * 2);
        WaP[l] = (_Float16*)take((size_t)FINP[l] * 2);
        bcomb[l] = (float*)take((size_t)FOUT[l] * 4);
    }

    // Weight prep (tiny)
    for (int l = 0; l < 4; ++l) {
        int tot = FINP[l] * FOUT[l];
        gnn_pack_w<<<(tot + 255) / 256, 256, 0, stream>>>(P.L[l].Ws, WsP[l], FINR[l], FINP[l], FOUT[l]);
        gnn_pack_w<<<(tot + 255) / 256, 256, 0, stream>>>(P.L[l].Wn, WnP[l], FINR[l], FINP[l], FOUT[l]);
        gnn_pack_att<<<1, 256, 0, stream>>>(P.L[l].Wa, WaP[l], FINR[l], FINP[l]);
        gnn_combine_bias<<<1, 256, 0, stream>>>(P.L[l].bs, P.L[l].bn, bcomb[l], FOUT[l]);
    }

    gnn_build_h0<<<MTOT / 256, 256, 0, stream>>>(P.x, hA);

    _Float16* cur = hA;
    _Float16* nxt = hB;
    for (int l = 0; l < 4; ++l) {
        gnn_att_score<<<MTOT / 256, 256, 0, stream>>>(cur, WaP[l], aj, FINP[l]);
        gnn_aggregate<<<MTOT / 4, 128, 0, stream>>>(cur, aj, hG, FINP[l]);
        dim3 gg(MTOT / 16 / 8), gb(256);
        switch (l) {
        case 0: gnn_gemm_ln<32, 64 ><<<gg, gb, 0, stream>>>(cur, hG, WsP[0], WnP[0], bcomb[0], P.L[0].lnw, P.L[0].lnb, nxt); break;
        case 1: gnn_gemm_ln<64, 128><<<gg, gb, 0, stream>>>(cur, hG, WsP[1], WnP[1], bcomb[1], P.L[1].lnw, P.L[1].lnb, nxt); break;
        case 2: gnn_gemm_ln<128,256><<<gg, gb, 0, stream>>>(cur, hG, WsP[2], WnP[2], bcomb[2], P.L[2].lnw, P.L[2].lnb, nxt); break;
        case 3: gnn_gemm_ln<256,128><<<gg, gb, 0, stream>>>(cur, hG, WsP[3], WnP[3], bcomb[3], P.L[3].lnw, P.L[3].lnb, nxt); break;
        }
        _Float16* tmp = cur; cur = nxt; nxt = tmp;
    }

    // cur: [MTOT,128] f16 final node features
    gnn_pool_score<<<MTOT / 4, 128, 0, stream>>>(cur, P.Wp1, P.bp1, P.Wp2, P.bp2, sc);
    gnn_pool_head<<<BATCH, 256, 0, stream>>>(cur, sc, P.Wc1, P.bc1, P.Wc2, P.bc2, (float*)d_out);
}
